// Attention_48610439856262
// MI455X (gfx1250) — matrix operand order repaired and checked by reading, not compiled, not run on hardware
//
#include <hip/hip_runtime.h>
#include <hip/hip_bf16.h>

typedef __bf16 bf16;
typedef __attribute__((ext_vector_type(4)))  __bf16 v4bf;
typedef __attribute__((ext_vector_type(8)))  __bf16 v8bf;
typedef __attribute__((ext_vector_type(16))) __bf16 v16bf;
typedef __attribute__((ext_vector_type(8)))  float  v8f;
typedef __attribute__((ext_vector_type(4)))  float  v4f;

union BF16Frag { v16bf v; v8bf h[2]; };

#define BM 128
#define BN 128
#define BK 32
#define LDS_AS 40   // padded LDS stride (bf16): conflict-free b128 frag loads
#define LDS_WS 40

// ---- CDNA5 async global->LDS path (guarded: falls back to sync staging) ----
#if __has_builtin(__builtin_amdgcn_global_load_async_to_lds_b128) && \
    __has_builtin(__builtin_amdgcn_s_wait_asynccnt)
#define USE_ASYNC 1
#else
#define USE_ASYNC 0
#endif

typedef int v4i128 __attribute__((vector_size(16)));
typedef __attribute__((address_space(1))) v4i128* gbl128;
typedef __attribute__((address_space(3))) v4i128* lds128;

__device__ __forceinline__ void cp16(bf16* lds, const bf16* g) {
#if USE_ASYNC
  __builtin_amdgcn_global_load_async_to_lds_b128((gbl128)g, (lds128)lds, 0, 0);
#else
  *(v8bf*)lds = *(const v8bf*)g;
#endif
}

template <int Nwait>
__device__ __forceinline__ void wait_cp() {
#if USE_ASYNC
  __builtin_amdgcn_s_wait_asynccnt(Nwait);
#endif
}

__device__ __forceinline__ v8f wmma_bf16(v16bf a, v16bf b, v8f c) {
  return __builtin_amdgcn_wmma_f32_16x16x32_bf16(
      false, a, false, b, (short)0, c, false, false);
}

// -----------------------------------------------------------------------------
// Prep: f32 -> bf16 elementwise (activations)
// -----------------------------------------------------------------------------
__global__ __launch_bounds__(256) void convert_bf16(
    const float* __restrict__ in, bf16* __restrict__ out, int n) {
  int i = (blockIdx.x * 256 + threadIdx.x) * 8;
  if (i + 8 <= n) {
    v4f a = *(const v4f*)(in + i);
    v4f b = *(const v4f*)(in + i + 4);
    v8bf pk;
    pk[0] = (bf16)a[0]; pk[1] = (bf16)a[1]; pk[2] = (bf16)a[2]; pk[3] = (bf16)a[3];
    pk[4] = (bf16)b[0]; pk[5] = (bf16)b[1]; pk[6] = (bf16)b[2]; pk[7] = (bf16)b[3];
    *(v8bf*)(out + i) = pk;
  }
}

// -----------------------------------------------------------------------------
// Prep: W[K][Nw] f32 -> Wt[Nw][K] bf16 (32x32 LDS tile transpose)
// -----------------------------------------------------------------------------
__global__ __launch_bounds__(256) void transpose_bf16(
    const float* __restrict__ W, bf16* __restrict__ Wt, int K, int Nw) {
  __shared__ float t[32][33];
  const int k0 = blockIdx.y * 32, n0 = blockIdx.x * 32;
  const int tr = threadIdx.x >> 3;        // 0..31
  const int tc = (threadIdx.x & 7) * 4;   // 0..28
  v4f v = *(const v4f*)(W + (size_t)(k0 + tr) * Nw + n0 + tc);
  t[tr][tc + 0] = v[0]; t[tr][tc + 1] = v[1];
  t[tr][tc + 2] = v[2]; t[tr][tc + 3] = v[3];
  __syncthreads();
  v4bf o;
  o[0] = (bf16)t[tc + 0][tr]; o[1] = (bf16)t[tc + 1][tr];
  o[2] = (bf16)t[tc + 2][tr]; o[3] = (bf16)t[tc + 3][tr];
  *(v4bf*)(Wt + (size_t)(n0 + tr) * K + k0 + tc) = o;
}

// -----------------------------------------------------------------------------
// Tiled GEMM: D[M,N] = A[M,K] bf16 * Wt[N,K] bf16 (weights pre-transposed)
// Double-buffered LDS staging via async global->LDS (when available).
// 256 threads = 8 wave32; each wave computes a 32x64 sub-tile (2x4 WMMA tiles).
// -----------------------------------------------------------------------------
template <bool ACC, bool BIAS, typename OT>
__global__ __launch_bounds__(256) void gemm_wmma(
    const bf16* __restrict__ A, int lda,
    const bf16* __restrict__ Wt,          // [N][K]
    OT* __restrict__ D, int ldd,
    const float* __restrict__ bias,
    int M, int K) {
  __shared__ bf16 As[2][BM * LDS_AS];
  __shared__ bf16 Ws[2][BN * LDS_WS];

  const int tid  = threadIdx.x;
  const int lane = tid & 31;
  const int wid  = tid >> 5;
  const int wm   = wid & 3;
  const int wn   = wid >> 2;
  const int half = lane >> 4;
  const int l16  = lane & 15;

  const int m0 = blockIdx.y * BM;
  const int n0 = blockIdx.x * BN;

  const int srow = tid >> 1;            // staging row 0..127
  const int sseg = (tid & 1) * 16;      // k segment 0 / 16

  v8f acc[2][4] = {};

  auto stage = [&](int k0, int buf) {
    const bf16* ga = A + (size_t)(m0 + srow) * lda + k0 + sseg;
    bf16* la = &As[buf][srow * LDS_AS + sseg];
    cp16(la, ga); cp16(la + 8, ga + 8);
    const bf16* gw = Wt + (size_t)(n0 + srow) * K + k0 + sseg;
    bf16* lw = &Ws[buf][srow * LDS_WS + sseg];
    cp16(lw, gw); cp16(lw + 8, gw + 8);
  };

  const int NS = K / BK;
  stage(0, 0);
  for (int j = 0; j < NS; ++j) {
    const int buf = j & 1;
    if (j + 1 < NS) {
      stage((j + 1) * BK, (j + 1) & 1);
      wait_cp<4>();   // stage j complete; stage j+1 (4 asyncs) in flight
    } else {
      wait_cp<0>();
    }
    __syncthreads();

    BF16Frag afr[2], bfr[4];
    for (int i = 0; i < 2; ++i) {
      // A layout: lanes<16 K={0..7,16..23}, lanes>=16 K={8..15,24..31}
      const bf16* p = &As[buf][(wm * 32 + i * 16 + l16) * LDS_AS + half * 8];
      afr[i].h[0] = *(const v8bf*)p;
      afr[i].h[1] = *(const v8bf*)(p + 16);
    }
    for (int jj = 0; jj < 4; ++jj) {
      // B layout: lanes<16 K={0..7,16..23}, lanes>=16 K={8..15,24..31}
      const bf16* p = &Ws[buf][(wn * 64 + jj * 16 + l16) * LDS_WS + half * 8];
      bfr[jj].h[0] = *(const v8bf*)p;
      bfr[jj].h[1] = *(const v8bf*)(p + 16);
    }
    for (int i = 0; i < 2; ++i)
      for (int jj = 0; jj < 4; ++jj)
        acc[i][jj] = wmma_bf16(afr[i].v, bfr[jj].v, acc[i][jj]);
    __syncthreads();  // buffer reuse safety (rewritten 2 stages later)
  }

  for (int i = 0; i < 2; ++i) {
    const int rbase = m0 + wm * 32 + i * 16 + half * 8;
    for (int jj = 0; jj < 4; ++jj) {
      const int col = n0 + wn * 64 + jj * 16 + l16;
      const float bv = BIAS ? bias[col] : 0.0f;
      for (int r = 0; r < 8; ++r) {
        float v = acc[i][jj][r] + bv;
        OT* op = D + (size_t)(rbase + r) * ldd + col;
        if constexpr (ACC) v += (float)*op;
        *op = (OT)v;
      }
    }
  }
}

// -----------------------------------------------------------------------------
// Flash attention, one (b,h,64-query-rows) block per workgroup (128 thr/4 waves)
// qkv: [B*N][3C] bf16 (K already s-biased); gate: [B*N][C] bf16
// out: [B*N][C] bf16 = softmax(QK^T*scale) V * gate
// -----------------------------------------------------------------------------
__global__ __launch_bounds__(128) void attn_wmma(
    const bf16* __restrict__ qkv,
    const bf16* __restrict__ gate,
    bf16* __restrict__ out) {
  constexpr int N = 2048, C = 1024, Dh = 64;
  constexpr int QC = 3 * C;
  constexpr float SCALE = 0.125f;  // 64^-0.5

  __shared__ bf16 Vt[64 * 72];        // V transposed: [d][key], padded
  __shared__ bf16 Ps[4][16 * 72];     // per-wave P scratch: [row][key]

  const int tid  = threadIdx.x;
  const int lane = tid & 31;
  const int wid  = tid >> 5;
  const int half = lane >> 4;
  const int l16  = lane & 15;

  const int blk = blockIdx.x;
  const int qb  = blk & 31;           // query block (N/64 = 32)
  const int bh  = blk >> 5;
  const int b   = bh >> 4;            // H = 16
  const int h   = bh & 15;

  const size_t rowBase = (size_t)b * N;
  const int    qRow0   = qb * 64 + wid * 16;

  // Q fragments (A layout), resident for entire kernel
  BF16Frag qf[2];
  {
    const bf16* qp = qkv + (rowBase + qRow0 + l16) * QC + h * Dh + half * 8;
    for (int s = 0; s < 2; ++s) {
      qf[s].h[0] = *(const v8bf*)(qp + s * 32);
      qf[s].h[1] = *(const v8bf*)(qp + s * 32 + 16);
    }
  }

  // V staging map: 4 adjacent keys x 8 d per thread -> packed ds_store_b64
  const int vkq = (tid & 15) * 4;     // key quad base
  const int vdc = (tid >> 4) * 8;     // d base

  v8f o[4] = {};
  float m[8], l[8];
  for (int r = 0; r < 8; ++r) { m[r] = -3.0e38f; l[r] = 0.0f; }

  for (int jb = 0; jb < N / 64; ++jb) {
    __syncthreads();  // prior P@V done reading Vt

    // ---- stage V block transposed into LDS: Vt[d][key] ----
    {
      const bf16* vp =
          qkv + (rowBase + jb * 64 + vkq) * QC + 2 * C + h * Dh + vdc;
      v8bf r0 = *(const v8bf*)(vp + 0 * QC);
      v8bf r1 = *(const v8bf*)(vp + 1 * QC);
      v8bf r2 = *(const v8bf*)(vp + 2 * QC);
      v8bf r3 = *(const v8bf*)(vp + 3 * QC);
      for (int e = 0; e < 8; ++e) {
        v4bf pk; pk[0] = r0[e]; pk[1] = r1[e]; pk[2] = r2[e]; pk[3] = r3[e];
        *(v4bf*)&Vt[(vdc + e) * 72 + vkq] = pk;
      }
    }

    // ---- S = Q K^T (key rows contiguous along D => direct B-layout loads) ----
    v8f st[4] = {};
    for (int t = 0; t < 4; ++t) {
      const bf16* kp =
          qkv + (rowBase + jb * 64 + t * 16 + l16) * QC + C + h * Dh + half * 8;
      for (int s = 0; s < 2; ++s) {
        BF16Frag kf;
        kf.h[0] = *(const v8bf*)(kp + s * 32);
        kf.h[1] = *(const v8bf*)(kp + s * 32 + 16);
        st[t] = wmma_bf16(qf[s].v, kf.v, st[t]);
      }
    }

    // ---- online softmax (reductions confined to each 16-lane half) ----
    for (int t = 0; t < 4; ++t)
      for (int r = 0; r < 8; ++r) st[t][r] *= SCALE;
    for (int r = 0; r < 8; ++r) {
      float mx = fmaxf(fmaxf(st[0][r], st[1][r]), fmaxf(st[2][r], st[3][r]));
      for (int msk = 8; msk >= 1; msk >>= 1)
        mx = fmaxf(mx, __shfl_xor(mx, msk, 32));
      const float mn = fmaxf(m[r], mx);
      const float al = __expf(m[r] - mn);
      m[r] = mn;
      float rs = 0.0f;
      for (int t = 0; t < 4; ++t) {
        const float p = __expf(st[t][r] - mn);
        st[t][r] = p;
        rs += p;
      }
      for (int msk = 8; msk >= 1; msk >>= 1) rs += __shfl_xor(rs, msk, 32);
      l[r] = l[r] * al + rs;
      for (int t = 0; t < 4; ++t) o[t][r] *= al;
    }

    // ---- C-layout -> A-layout for P via per-wave LDS scratch ----
    bf16* ps = &Ps[wid][0];
    for (int t = 0; t < 4; ++t)
      for (int r = 0; r < 8; ++r)
        ps[(r + half * 8) * 72 + t * 16 + l16] = (bf16)st[t][r];

    __syncthreads();  // Vt ready

    // ---- O += P @ V ----
    for (int s = 0; s < 2; ++s) {
      BF16Frag pf;
      const bf16* pp = ps + l16 * 72 + s * 32 + half * 8;
      pf.h[0] = *(const v8bf*)pp;
      pf.h[1] = *(const v8bf*)(pp + 16);
      for (int t = 0; t < 4; ++t) {
        BF16Frag vf;
        const bf16* vp2 = &Vt[(t * 16 + l16) * 72 + s * 32 + half * 8];
        vf.h[0] = *(const v8bf*)vp2;
        vf.h[1] = *(const v8bf*)(vp2 + 16);
        o[t] = wmma_bf16(pf.v, vf.v, o[t]);
      }
    }
  }

  // ---- epilogue: normalize, apply gate, store bf16 ----
  for (int t = 0; t < 4; ++t) {
    const int col = h * Dh + t * 16 + l16;
    for (int r = 0; r < 8; ++r) {
      const size_t row = rowBase + qRow0 + half * 8 + r;
      const float g = (float)gate[row * C + col];
      out[row * C + col] = (bf16)(o[t][r] * (1.0f / l[r]) * g);
    }
  }
}

// -----------------------------------------------------------------------------
extern "C" void kernel_launch(void* const* d_in, const int* in_sizes, int n_in,
                              void* d_out, int out_size, void* d_ws, size_t ws_size,
                              hipStream_t stream) {
  const float* x      = (const float*)d_in[0];
  const float* e      = (const float*)d_in[1];
  const float* W_qkv  = (const float*)d_in[2];
  const float* W_s    = (const float*)d_in[3];
  const float* W_gate = (const float*)d_in[4];
  const float* W_proj = (const float*)d_in[5];
  const float* b_proj = (const float*)d_in[6];
  float* outp = (float*)d_out;

  const int B = 2, N = 2048, C = 1024;
  const int M = B * N;  // 4096
  const size_t MC = (size_t)M * C;

  bf16* p = (bf16*)d_ws;
  bf16* xb    = p; p += MC;            // [M][C]
  bf16* eb    = p; p += MC;            // [M][C]
  bf16* WqkvT = p; p += (size_t)3 * C * C;  // [3C][C]
  bf16* WsT   = p; p += (size_t)C * C;      // [C][C]
  bf16* WgT   = p; p += (size_t)C * C;      // [C][C]
  bf16* WpT   = p; p += (size_t)C * C;      // [C][C]
  bf16* qkvb  = p; p += (size_t)M * 3 * C;  // [M][3C]
  bf16* gateb = p; p += MC;            // [M][C]
  bf16* attnb = p;                     // [M][C]

  // ---- prep: convert activations, convert+transpose weights ----
  convert_bf16<<<dim3((int)(MC / (256 * 8))), 256, 0, stream>>>(x, xb, (int)MC);
  convert_bf16<<<dim3((int)(MC / (256 * 8))), 256, 0, stream>>>(e, eb, (int)MC);
  transpose_bf16<<<dim3(3 * C / 32, C / 32), 256, 0, stream>>>(W_qkv, WqkvT, C, 3 * C);
  transpose_bf16<<<dim3(C / 32, C / 32), 256, 0, stream>>>(W_s,    WsT, C, C);
  transpose_bf16<<<dim3(C / 32, C / 32), 256, 0, stream>>>(W_gate, WgT, C, C);
  transpose_bf16<<<dim3(C / 32, C / 32), 256, 0, stream>>>(W_proj, WpT, C, C);

  // 1) qkv = x @ W_qkv
  gemm_wmma<false, false, bf16><<<dim3(3 * C / BN, M / BM), 256, 0, stream>>>(
      xb, C, WqkvT, qkvb, 3 * C, nullptr, M, C);
  // 2) K slice += e @ W_s (fuses s-bias into keys)
  gemm_wmma<true, false, bf16><<<dim3(C / BN, M / BM), 256, 0, stream>>>(
      eb, C, WsT, qkvb + C, 3 * C, nullptr, M, C);
  // 3) gate = x @ W_gate
  gemm_wmma<false, false, bf16><<<dim3(C / BN, M / BM), 256, 0, stream>>>(
      xb, C, WgT, gateb, C, nullptr, M, C);
  // 4) flash attention + gate
  attn_wmma<<<dim3(B * 16 * (N / 64)), 128, 0, stream>>>(qkvb, gateb, attnb);
  // 5) out = (attn*gate) @ W_proj + b_proj
  gemm_wmma<false, true, float><<<dim3(C / BN, M / BM), 256, 0, stream>>>(
      attnb, C, WpT, outp, C, b_proj, M, C);
}
